// MultiHeadAttention4D_51711406244114
// MI455X (gfx1250) — matrix operand order repaired and checked by reading, not compiled, not run on hardware
//
#include <hip/hip_runtime.h>
#include <hip/hip_bf16.h>

typedef __attribute__((ext_vector_type(16))) _Float16 v16h;
typedef __attribute__((ext_vector_type(8)))  _Float16 v8h;
typedef __attribute__((ext_vector_type(8)))  float    v8f;
typedef __attribute__((ext_vector_type(4)))  float    v4f;

constexpr int    kE   = 512;                 // embed dim
constexpr int    kS   = 512;                 // sequence
constexpr int    kBC  = 64;                  // B*C
constexpr size_t kSLAB = (size_t)kS * kE;            // 262144 elems per (b,c) slab
constexpr size_t kWT   = (size_t)kE * kE;            // 262144 elems per weight
constexpr size_t kQKV  = (size_t)kBC * kSLAB;        // 16,777,216 elems
constexpr int    kSCST = 520;                // padded LDS score row stride (floats, 16B-aligned)

// ---------------------------------------------------------------------------
// Shared GEMM core: one wave computes a 16x16 tile of (512x512) @ Wt + bias.
// Wt is pre-swizzled: element (k, n) lives at Wt[(n/16)*512*16 + k*16 + (n%16)]
// so each k row of a B tile is 16 contiguous f16 (32B) across lanes, coalesced.
// ---------------------------------------------------------------------------
template <bool A_F32, bool OUT_F16>
__device__ __forceinline__ void gemm512(const void* __restrict__ Abc,
                                        const _Float16* __restrict__ Wt,
                                        const float* __restrict__ bias,
                                        void* __restrict__ Obc) {
  const int lane = threadIdx.x & 31;
  const int wv   = threadIdx.x >> 5;
  const int t    = blockIdx.x * 8 + wv;     // 1024 tiles per (b,c)
  const int m0   = (t >> 5) << 4;
  const int n0   = (t & 31) << 4;
  const int arow = m0 + (lane & 15);
  const int ksel = (lane >> 4) << 3;        // ISA 16-bit A layout: lanes>=16 start at K+8
  const _Float16* __restrict__ Bt = Wt + (size_t)(n0 >> 4) * (kE * 16);

  v8f acc = {};
#pragma unroll 4
  for (int kk = 0; kk < kE; kk += 32) {
    v16h a;
    if constexpr (A_F32) {
      const float* xr = (const float*)Abc + (size_t)arow * kE + kk + ksel;
      v4f x0 = *(const v4f*)(xr);
      v4f x1 = *(const v4f*)(xr + 4);
      v4f x2 = *(const v4f*)(xr + 16);
      v4f x3 = *(const v4f*)(xr + 20);
#pragma unroll
      for (int i = 0; i < 4; ++i) {
        a[i]      = (_Float16)x0[i];
        a[4 + i]  = (_Float16)x1[i];
        a[8 + i]  = (_Float16)x2[i];
        a[12 + i] = (_Float16)x3[i];
      }
    } else {
      const _Float16* xr = (const _Float16*)Abc + (size_t)arow * kE + kk + ksel;
      v8h lo = *(const v8h*)(xr);
      v8h hi = *(const v8h*)(xr + 16);
#pragma unroll
      for (int i = 0; i < 8; ++i) { a[i] = lo[i]; a[8 + i] = hi[i]; }
    }
    v16h b;
#pragma unroll
    for (int s = 0; s < 16; ++s) b[s] = Bt[(size_t)(kk + ksel + s + (s & 8)) * 16 + (lane & 15)];
    if (kk + 32 < kE) __builtin_prefetch(Bt + (size_t)(kk + 32 + lane) * 16, 0, 0);
    acc = __builtin_amdgcn_wmma_f32_16x16x32_f16(false, a, false, b, (short)0, acc,
                                                 false, false);
  }

  const int   ocol  = n0 + (lane & 15);
  const int   orow0 = m0 + ((lane >> 4) << 3);
  const float bb    = bias[ocol];
#pragma unroll
  for (int i = 0; i < 8; ++i) {
    float r = acc[i] + bb;
    if constexpr (OUT_F16) {
      ((_Float16*)Obc)[(size_t)(orow0 + i) * kE + ocol] = (_Float16)r;
    } else {
      __builtin_nontemporal_store(r, (float*)Obc + (size_t)(orow0 + i) * kE + ocol);
    }
  }
}

// ---------------------------------------------------------------------------
// Kernel 1: convert+swizzle the four 512x512 fp32 weights to f16 tile layout.
// y = x @ W.T  =>  B(k, n) = W(n, k).
// ---------------------------------------------------------------------------
__global__ __launch_bounds__(256) void wt_kernel(const float* __restrict__ wq,
                                                 const float* __restrict__ wk,
                                                 const float* __restrict__ wv,
                                                 const float* __restrict__ wo,
                                                 _Float16* __restrict__ wt) {
  const int w = blockIdx.y;
  const float* W = (w == 0) ? wq : (w == 1) ? wk : (w == 2) ? wv : wo;
  const int idx = blockIdx.x * 256 + threadIdx.x;   // coalesced read of W
  const int n   = idx >> 9;                          // output-feature row of W
  const int k   = idx & 511;
  wt[(size_t)w * kWT + ((size_t)(n >> 4) * kE + k) * 16 + (n & 15)] =
      (_Float16)W[idx];
}

// ---------------------------------------------------------------------------
// Kernel 2: Q/K/V projections (fp32 in, f16 out), one grid.z per matrix.
// ---------------------------------------------------------------------------
__global__ __launch_bounds__(256) void proj_qkv_kernel(
    const float* __restrict__ q, const float* __restrict__ k,
    const float* __restrict__ v, const _Float16* __restrict__ wt,
    const float* __restrict__ bq, const float* __restrict__ bk,
    const float* __restrict__ bv, _Float16* __restrict__ qkv) {
  const int z  = blockIdx.z;
  const int bc = blockIdx.y;
  const float* X;
  const float* bias;
  if (z == 0)      { X = q; bias = bq; }
  else if (z == 1) { X = k; bias = bk; }
  else             { X = v; bias = bv; }
  gemm512<true, true>(X + (size_t)bc * kSLAB, wt + (size_t)z * kWT, bias,
                      qkv + ((size_t)z * kBC + bc) * kSLAB);
}

// ---------------------------------------------------------------------------
// Kernel 3: attention. One block (4 waves) = one (b,h,c) x 16-query strip.
//   scores -> LDS, softmax, coalesced NT store of attn_weights, context GEMM.
// ---------------------------------------------------------------------------
__global__ __launch_bounds__(128) void attn_kernel(
    const _Float16* __restrict__ Qh, const _Float16* __restrict__ Kh,
    const _Float16* __restrict__ Vh, float* __restrict__ attn_out,
    _Float16* __restrict__ ctx) {
  __shared__ float sc[16 * kSCST];     // 16 query rows x 512 keys, padded
  __shared__ float red[16][8];
  __shared__ float rowstat[16];

  const int tid  = threadIdx.x;
  const int lane = tid & 31;
  const int wv   = tid >> 5;
  const int y    = blockIdx.y;               // (b*H + h)*C + c
  const int b    = y >> 7;
  const int h    = (y >> 4) & 7;
  const int c    = y & 15;
  const int q0   = blockIdx.x << 4;

  const size_t base = (((size_t)b * 16 + c) * kS) * kE + (size_t)h * 64;
  const _Float16* Qb = Qh + base;
  const _Float16* Kb = Kh + base;
  const _Float16* Vb = Vh + base;

  const int arow = q0 + (lane & 15);
  const int ksel = (lane >> 4) << 3;
  const int lr0  = (lane >> 4) << 3;

  // ---- phase 1: raw scores (q strip x all 512 keys), 8 n-tiles per wave ----
  for (int nt = wv; nt < 32; nt += 4) {
    const int n0 = nt << 4;
    v8f acc = {};
#pragma unroll
    for (int kk = 0; kk < 64; kk += 32) {
      const _Float16* qr = Qb + (size_t)arow * kE + kk + ksel;
      v8h lo = *(const v8h*)(qr);
      v8h hi = *(const v8h*)(qr + 16);
      v16h a;
#pragma unroll
      for (int i = 0; i < 8; ++i) { a[i] = lo[i]; a[8 + i] = hi[i]; }
      // B = K^T tile: lane%16 indexes the key position,
      // elements index head-dim in the same k order as A.
      const _Float16* kc = Kb + (size_t)(n0 + (lane & 15)) * kE + kk + ksel;
      v16h bm;
#pragma unroll
      for (int t2 = 0; t2 < 16; ++t2) bm[t2] = kc[t2 + (t2 & 8)];
      acc = __builtin_amdgcn_wmma_f32_16x16x32_f16(false, a, false, bm, (short)0,
                                                   acc, false, false);
    }
    const int lcol = n0 + (lane & 15);
#pragma unroll
    for (int i = 0; i < 8; ++i) sc[(lr0 + i) * kSCST + lcol] = acc[i];
  }
  __syncthreads();

  // ---- phase 2: softmax over 512 keys per row (scale folded into exp) ----
  const int r  = tid >> 3;
  const int s8 = tid & 7;
  float* srow = sc + r * kSCST;
  float mx = -1e30f;
#pragma unroll 8
  for (int j = 0; j < 64; ++j) mx = fmaxf(mx, srow[s8 + (j << 3)]);
  red[r][s8] = mx;
  __syncthreads();
  if (tid < 16) {
    float m2 = red[tid][0];
#pragma unroll
    for (int j = 1; j < 8; ++j) m2 = fmaxf(m2, red[tid][j]);
    rowstat[tid] = m2;
  }
  __syncthreads();
  const float m = rowstat[r];
  const float scale = 0.125f;   // 1/sqrt(64)
  float sum = 0.f;
#pragma unroll 8
  for (int j = 0; j < 64; ++j) {
    const int ci = s8 + (j << 3);
    const float e = __expf((srow[ci] - m) * scale);
    srow[ci] = e;
    sum += e;
  }
  red[r][s8] = sum;
  __syncthreads();
  if (tid < 16) {
    float s2 = 0.f;
#pragma unroll
    for (int j = 0; j < 8; ++j) s2 += red[tid][j];
    rowstat[tid] = 1.0f / s2;
  }
  __syncthreads();

  // ---- phase 3: normalize in LDS + coalesced nontemporal store to d_out ----
  const size_t attn_base = ((size_t)y * kS + q0) * kS;
#pragma unroll 4
  for (int j = 0; j < 16; ++j) {
    const int idx4 = tid + (j << 7);         // 2048 float4s
    const int rr   = idx4 >> 7;
    const int c4   = idx4 & 127;
    v4f e4 = *(v4f*)(sc + rr * kSCST + (c4 << 2));
    e4 *= rowstat[rr];
    *(v4f*)(sc + rr * kSCST + (c4 << 2)) = e4;
    __builtin_nontemporal_store(
        e4, (v4f*)(attn_out + attn_base + (size_t)rr * kS + (c4 << 2)));
  }
  __syncthreads();

  // ---- phase 4: context = attn(16x512) @ V(512x64); 1 n-tile per wave ----
  {
    const int n0 = wv << 4;
    v8f acc = {};
    for (int kk = 0; kk < kS; kk += 32) {
      const float* ar = sc + (lane & 15) * kSCST + kk + ksel;
      v4f x0 = *(const v4f*)(ar);
      v4f x1 = *(const v4f*)(ar + 4);
      v4f x2 = *(const v4f*)(ar + 16);
      v4f x3 = *(const v4f*)(ar + 20);
      v16h a;
#pragma unroll
      for (int i = 0; i < 4; ++i) {
        a[i]      = (_Float16)x0[i];
        a[4 + i]  = (_Float16)x1[i];
        a[8 + i]  = (_Float16)x2[i];
        a[12 + i] = (_Float16)x3[i];
      }
      v16h bmat;
#pragma unroll
      for (int s = 0; s < 16; ++s) bmat[s] = Vb[(size_t)(kk + ksel + s + (s & 8)) * kE + n0 + (lane & 15)];
      acc = __builtin_amdgcn_wmma_f32_16x16x32_f16(false, a, false, bmat, (short)0,
                                                   acc, false, false);
    }
    const int ocol  = n0 + (lane & 15);
    const int orow0 = q0 + lr0;
#pragma unroll
    for (int i = 0; i < 8; ++i)
      ctx[base + (size_t)(orow0 + i) * kE + ocol] = (_Float16)acc[i];
  }
}

// ---------------------------------------------------------------------------
// Kernel 4: output projection (f16 context in, fp32 out).
// ---------------------------------------------------------------------------
__global__ __launch_bounds__(256) void out_proj_kernel(
    const _Float16* __restrict__ ctx, const _Float16* __restrict__ wto,
    const float* __restrict__ bo, float* __restrict__ out) {
  const int bc = blockIdx.y;
  gemm512<false, false>(ctx + (size_t)bc * kSLAB, wto, bo,
                        out + (size_t)bc * kSLAB);
}

extern "C" void kernel_launch(void* const* d_in, const int* in_sizes, int n_in,
                              void* d_out, int out_size, void* d_ws,
                              size_t ws_size, hipStream_t stream) {
  const float* q  = (const float*)d_in[0];
  const float* k  = (const float*)d_in[1];
  const float* v  = (const float*)d_in[2];
  const float* Wq = (const float*)d_in[3];
  const float* bq = (const float*)d_in[4];
  const float* Wk = (const float*)d_in[5];
  const float* bk = (const float*)d_in[6];
  const float* Wv = (const float*)d_in[7];
  const float* bv = (const float*)d_in[8];
  const float* Wo = (const float*)d_in[9];
  const float* bo = (const float*)d_in[10];

  float* out  = (float*)d_out;
  float* attn = out + (size_t)kBC * kSLAB;       // second tuple element

  _Float16* wt  = (_Float16*)d_ws;               // 4 swizzled f16 weights
  _Float16* qkv = wt + 4 * kWT;                  // Q,K,V f16 [z][b][c][s][e]
  _Float16* ctx = qkv + 3 * kQKV;                // context f16 [b][c][s][e]

  wt_kernel<<<dim3(1024, 4), 256, 0, stream>>>(Wq, Wk, Wv, Wo, wt);
  proj_qkv_kernel<<<dim3(128, 64, 3), 256, 0, stream>>>(q, k, v, wt, bq, bk, bv,
                                                        qkv);
  attn_kernel<<<dim3(32, 512), 128, 0, stream>>>(qkv, qkv + kQKV, qkv + 2 * kQKV,
                                                 attn, ctx);
  out_proj_kernel<<<dim3(128, 64), 256, 0, stream>>>(ctx, wt + 3 * kWT, bo, out);
}
